// RGCNModel_18631568130925
// MI455X (gfx1250) — compile-verified, matrix operand order checked by reading, not run on hardware
//
#include <hip/hip_runtime.h>
#include <hip/hip_bf16.h>

typedef __attribute__((ext_vector_type(16))) _Float16     v16h;
typedef __attribute__((ext_vector_type(8)))  float        v8f;
typedef __attribute__((ext_vector_type(8)))  unsigned int v8u;

#define COL_STRIDE 40   // halves per weight column in LDS (80 B: 16B-aligned, low conflicts)

// ---------------------------------------------------------------------------
// x[i,k] = emb[feats[i], k]
__global__ void gather_x_kernel(const float* __restrict__ emb,
                                const int* __restrict__ feats,
                                float* __restrict__ x, int total) {
    int i = blockIdx.x * blockDim.x + threadIdx.x;
    if (i < total) {
        int n = i >> 5, k = i & 31;
        x[i] = emb[((size_t)feats[n] << 5) + k];
    }
}

// buf[i] = bias[i & mask]
__global__ void init_bias_kernel(float* __restrict__ buf,
                                 const float* __restrict__ bias,
                                 int mask, int total) {
    int i = blockIdx.x * blockDim.x + threadIdx.x;
    if (i < total) buf[i] = bias[i & mask];
}

// Load one B fragment (16 halves) from LDS at half-element offset `ofs`:
// two aligned 16B ds_load_b128. `ofs` must carry the per-iteration opaque
// component so these loads cannot be hoisted (and then spilled) by LICM.
__device__ __forceinline__ v16h load_bfrag(const _Float16* Wlds, int ofs) {
    union { uint4 q[2]; v16h h; } bu;
    bu.q[0] = *(const uint4*)(Wlds + ofs);
    bu.q[1] = *(const uint4*)(Wlds + ofs + 16);
    return bu.h;
}

// ---------------------------------------------------------------------------
// Layer 1: for 16-edge tiles, A[16x32] = norm * x[src]; accumulate over r:
//   D += mask_r(A) @ W1[r]  (two 16x16 N-tiles since HID=32), scatter to hbuf.
__global__ void rgcn_layer1_kernel(const float* __restrict__ x,
                                   const int*   __restrict__ src,
                                   const int*   __restrict__ dst,
                                   const int*   __restrict__ etype,
                                   const float* __restrict__ enorm,
                                   const float* __restrict__ W1,   // [8][32][32]
                                   float*       __restrict__ hbuf, // [N][32]
                                   int E) {
    // Stage W1^T as f16 in LDS: Wlds[(r*32 + j)*COL_STRIDE + k] = W1[r][k][j]
    __shared__ _Float16 Wlds[8 * 32 * COL_STRIDE];
    for (int idx = threadIdx.x; idx < 8 * 32 * 32; idx += blockDim.x) {
        int j = idx & 31, k = (idx >> 5) & 31, r = idx >> 10;   // global reads coalesced in j
        Wlds[(r * 32 + j) * COL_STRIDE + k] = (_Float16)W1[r * 1024 + k * 32 + j];
    }
    __syncthreads();

    const int lane  = threadIdx.x & 31;
    const int hi    = lane >> 4;        // lane half: selects K sub-range
    const int lm    = lane & 15;        // row (A) / column (B,D) index
    const int kbase = hi ? 8 : 0;

    const int waves  = (int)(gridDim.x * (blockDim.x >> 5));
    const int gwave  = blockIdx.x * (blockDim.x >> 5) + (threadIdx.x >> 5);
    const int ntiles = (E + 15) >> 4;

    for (int tile = gwave; tile < ntiles; tile += waves) {
        const int  erow = tile * 16 + lm;       // edge owned by this lane pair
        const bool ein  = erow < E;
        int   s  = ein ? src[erow]   : 0;
        int   t  = ein ? etype[erow] : -1;
        float nr = ein ? enorm[erow] : 0.0f;
        int   dv = ein ? dst[erow]   : -1;

        // Gather A row (this lane's K sub-range), pre-scaled by edge_norm.
        const float* xr = x + ((size_t)s << 5) + kbase;
        float4 a0 = *(const float4*)(xr + 0);
        float4 a1 = *(const float4*)(xr + 4);
        float4 a2 = *(const float4*)(xr + 16);
        float4 a3 = *(const float4*)(xr + 20);
        v16h A;
        A[0]  = (_Float16)(a0.x * nr); A[1]  = (_Float16)(a0.y * nr);
        A[2]  = (_Float16)(a0.z * nr); A[3]  = (_Float16)(a0.w * nr);
        A[4]  = (_Float16)(a1.x * nr); A[5]  = (_Float16)(a1.y * nr);
        A[6]  = (_Float16)(a1.z * nr); A[7]  = (_Float16)(a1.w * nr);
        A[8]  = (_Float16)(a2.x * nr); A[9]  = (_Float16)(a2.y * nr);
        A[10] = (_Float16)(a2.z * nr); A[11] = (_Float16)(a2.w * nr);
        A[12] = (_Float16)(a3.x * nr); A[13] = (_Float16)(a3.y * nr);
        A[14] = (_Float16)(a3.z * nr); A[15] = (_Float16)(a3.w * nr);

        // Opaque zero, redefined every iteration: blocks LICM of the B-frag
        // LDS loads (which previously re-materialized 128 VGPRs and spilled).
        int wofs = 0;
        asm volatile("" : "+v"(wofs));

        v8u au = __builtin_bit_cast(v8u, A);
        v8f D0 = {}; v8f D1 = {};
#pragma unroll
        for (int r = 0; r < 8; ++r) {
            bool m = (t == r);               // zero rows not of relation r
            v8u aru;
#pragma unroll
            for (int i = 0; i < 8; ++i) aru[i] = m ? au[i] : 0u;
            v16h Ar = __builtin_bit_cast(v16h, aru);
            v16h B0 = load_bfrag(Wlds, wofs + (r * 32 + lm)      * COL_STRIDE + kbase);
            v16h B1 = load_bfrag(Wlds, wofs + (r * 32 + lm + 16) * COL_STRIDE + kbase);
            D0 = __builtin_amdgcn_wmma_f32_16x16x32_f16(false, Ar, false, B0,
                                                        (short)0, D0, false, false);
            D1 = __builtin_amdgcn_wmma_f32_16x16x32_f16(false, Ar, false, B1,
                                                        (short)0, D1, false, false);
        }

        // Scatter: lane holds column lm (+16), rows j + 8*hi of the tile.
#pragma unroll
        for (int j = 0; j < 8; ++j) {
            int row = j + 8 * hi;
            int d   = __shfl(dv, row, 32);
            if (d >= 0) {
                atomicAdd(&hbuf[((size_t)d << 5) + lm],      D0[j]);
                atomicAdd(&hbuf[((size_t)d << 5) + lm + 16], D1[j]);
            }
        }
    }
}

// ---------------------------------------------------------------------------
// Layer 2: A[16x32] = norm * relu(h[src]); D += mask_r(A) @ W2[r] (OUT=16).
__global__ void rgcn_layer2_kernel(const float* __restrict__ hbuf,
                                   const int*   __restrict__ src,
                                   const int*   __restrict__ dst,
                                   const int*   __restrict__ etype,
                                   const float* __restrict__ enorm,
                                   const float* __restrict__ W2,   // [8][32][16]
                                   float*       __restrict__ obuf, // [N][16]
                                   int E) {
    // Stage W2^T as f16 in LDS: Wlds[(r*16 + j)*COL_STRIDE + k] = W2[r][k][j]
    __shared__ _Float16 Wlds[8 * 16 * COL_STRIDE];
    for (int idx = threadIdx.x; idx < 8 * 16 * 32; idx += blockDim.x) {
        int j = idx & 15, k = (idx >> 4) & 31, r = idx >> 9;
        Wlds[(r * 16 + j) * COL_STRIDE + k] = (_Float16)W2[r * 512 + k * 16 + j];
    }
    __syncthreads();

    const int lane  = threadIdx.x & 31;
    const int hi    = lane >> 4;
    const int lm    = lane & 15;
    const int kbase = hi ? 8 : 0;

    const int waves  = (int)(gridDim.x * (blockDim.x >> 5));
    const int gwave  = blockIdx.x * (blockDim.x >> 5) + (threadIdx.x >> 5);
    const int ntiles = (E + 15) >> 4;

    for (int tile = gwave; tile < ntiles; tile += waves) {
        const int  erow = tile * 16 + lm;
        const bool ein  = erow < E;
        int   s  = ein ? src[erow]   : 0;
        int   t  = ein ? etype[erow] : -1;
        float nr = ein ? enorm[erow] : 0.0f;
        int   dv = ein ? dst[erow]   : -1;

        const float* xr = hbuf + ((size_t)s << 5) + kbase;
        float4 a0 = *(const float4*)(xr + 0);
        float4 a1 = *(const float4*)(xr + 4);
        float4 a2 = *(const float4*)(xr + 16);
        float4 a3 = *(const float4*)(xr + 20);
        v16h A;  // ReLU fused into the gather
        A[0]  = (_Float16)(fmaxf(a0.x, 0.f) * nr); A[1]  = (_Float16)(fmaxf(a0.y, 0.f) * nr);
        A[2]  = (_Float16)(fmaxf(a0.z, 0.f) * nr); A[3]  = (_Float16)(fmaxf(a0.w, 0.f) * nr);
        A[4]  = (_Float16)(fmaxf(a1.x, 0.f) * nr); A[5]  = (_Float16)(fmaxf(a1.y, 0.f) * nr);
        A[6]  = (_Float16)(fmaxf(a1.z, 0.f) * nr); A[7]  = (_Float16)(fmaxf(a1.w, 0.f) * nr);
        A[8]  = (_Float16)(fmaxf(a2.x, 0.f) * nr); A[9]  = (_Float16)(fmaxf(a2.y, 0.f) * nr);
        A[10] = (_Float16)(fmaxf(a2.z, 0.f) * nr); A[11] = (_Float16)(fmaxf(a2.w, 0.f) * nr);
        A[12] = (_Float16)(fmaxf(a3.x, 0.f) * nr); A[13] = (_Float16)(fmaxf(a3.y, 0.f) * nr);
        A[14] = (_Float16)(fmaxf(a3.z, 0.f) * nr); A[15] = (_Float16)(fmaxf(a3.w, 0.f) * nr);

        int wofs = 0;
        asm volatile("" : "+v"(wofs));

        v8u au = __builtin_bit_cast(v8u, A);
        v8f D = {};
#pragma unroll
        for (int r = 0; r < 8; ++r) {
            bool m = (t == r);
            v8u aru;
#pragma unroll
            for (int i = 0; i < 8; ++i) aru[i] = m ? au[i] : 0u;
            v16h Ar = __builtin_bit_cast(v16h, aru);
            v16h B = load_bfrag(Wlds, wofs + (r * 16 + lm) * COL_STRIDE + kbase);
            D = __builtin_amdgcn_wmma_f32_16x16x32_f16(false, Ar, false, B,
                                                       (short)0, D, false, false);
        }

#pragma unroll
        for (int j = 0; j < 8; ++j) {
            int row = j + 8 * hi;
            int d   = __shfl(dv, row, 32);
            if (d >= 0) atomicAdd(&obuf[((size_t)d << 4) + lm], D[j]);
        }
    }
}

// ---------------------------------------------------------------------------
// Row-wise softmax over 16 logits.
__global__ void softmax16_kernel(const float* __restrict__ obuf,
                                 float* __restrict__ out, int N) {
    int n = blockIdx.x * blockDim.x + threadIdx.x;
    if (n >= N) return;
    const float4* rp = (const float4*)(obuf + ((size_t)n << 4));
    float4 v0 = rp[0], v1 = rp[1], v2 = rp[2], v3 = rp[3];
    float v[16] = {v0.x, v0.y, v0.z, v0.w, v1.x, v1.y, v1.z, v1.w,
                   v2.x, v2.y, v2.z, v2.w, v3.x, v3.y, v3.z, v3.w};
    float mx = v[0];
#pragma unroll
    for (int i = 1; i < 16; ++i) mx = fmaxf(mx, v[i]);
    float s = 0.f;
#pragma unroll
    for (int i = 0; i < 16; ++i) { v[i] = __expf(v[i] - mx); s += v[i]; }
    float inv = 1.0f / s;
    float4* op = (float4*)(out + ((size_t)n << 4));
    op[0] = make_float4(v[0] * inv,  v[1] * inv,  v[2] * inv,  v[3] * inv);
    op[1] = make_float4(v[4] * inv,  v[5] * inv,  v[6] * inv,  v[7] * inv);
    op[2] = make_float4(v[8] * inv,  v[9] * inv,  v[10] * inv, v[11] * inv);
    op[3] = make_float4(v[12] * inv, v[13] * inv, v[14] * inv, v[15] * inv);
}

// ---------------------------------------------------------------------------
extern "C" void kernel_launch(void* const* d_in, const int* in_sizes, int n_in,
                              void* d_out, int out_size, void* d_ws, size_t ws_size,
                              hipStream_t stream) {
    (void)n_in; (void)out_size; (void)ws_size;
    const int*   feats = (const int*)  d_in[0];
    const int*   src   = (const int*)  d_in[1];
    const int*   dst   = (const int*)  d_in[2];
    const int*   etype = (const int*)  d_in[3];
    const float* enorm = (const float*)d_in[4];
    const float* emb   = (const float*)d_in[5];
    const float* W1    = (const float*)d_in[6];
    const float* b1    = (const float*)d_in[7];
    const float* W2    = (const float*)d_in[8];
    const float* b2    = (const float*)d_in[9];
    float*       out   = (float*)d_out;

    const int N = in_sizes[0];
    const int E = in_sizes[1];

    // Workspace: x[N*32] | hbuf[N*32] | obuf[N*16]  (~32 MB @ N=100k)
    float* x    = (float*)d_ws;
    float* hbuf = x    + (size_t)N * 32;
    float* obuf = hbuf + (size_t)N * 32;

    const int ntiles = (E + 15) >> 4;
    int blocks = (ntiles + 7) / 8;          // 8 waves/block, 1 tile/wave min
    if (blocks > 2048) blocks = 2048;       // grid-stride beyond this

    gather_x_kernel<<<(N * 32 + 255) / 256, 256, 0, stream>>>(emb, feats, x, N * 32);
    init_bias_kernel<<<(N * 32 + 255) / 256, 256, 0, stream>>>(hbuf, b1, 31, N * 32);
    rgcn_layer1_kernel<<<blocks, 256, 0, stream>>>(x, src, dst, etype, enorm, W1, hbuf, E);
    init_bias_kernel<<<(N * 16 + 255) / 256, 256, 0, stream>>>(obuf, b2, 15, N * 16);
    rgcn_layer2_kernel<<<blocks, 256, 0, stream>>>(hbuf, src, dst, etype, enorm, W2, obuf, E);
    softmax16_kernel<<<(N + 255) / 256, 256, 0, stream>>>(obuf, out, N);
}
